// GKAMP_9929964388486
// MI455X (gfx1250) — matrix operand order repaired and checked by reading, not compiled, not run on hardware
//
#include <hip/hip_runtime.h>
#include <hip/hip_bf16.h>
#include <math.h>

// NodeFormer-style kernelized attention for MI455X (gfx1250, wave32, WMMA).
// B=1, N=20000, IN_C=256, OUT_C(D)=64, H=4, M=30 (padded to 32), E=320000.

typedef __attribute__((ext_vector_type(16))) __bf16 v16bf;
typedef __attribute__((ext_vector_type(8)))  __bf16 v8bf;
typedef __attribute__((ext_vector_type(4)))  __bf16 v4bf;
typedef __attribute__((ext_vector_type(8)))  float  v8f;

#define H_   4
#define D_   64
#define INC_ 256
#define M_   30
#define MP_  32   // padded feature count

// ---- order-preserving float<->uint encoding for atomicMax on floats ----
static __device__ __forceinline__ unsigned fenc(float f) {
  unsigned u = __float_as_uint(f);
  return (u & 0x80000000u) ? ~u : (u | 0x80000000u);
}
static __device__ __forceinline__ float fdec(unsigned u) {
  return (u & 0x80000000u) ? __uint_as_float(u ^ 0x80000000u)
                           : __uint_as_float(~u);
}

// ---- WMMA fragment loaders (wave32; contiguous k order, same in A and B) ----
// A: 16x32 (MxK) bf16. Lane holds row (lane&15); lanes 0-15 cover K
// {0..15}, lanes 16-31 cover {16..31} (kh = lane>>4).

// f32 source with on-the-fly convert (+scale folded in)
static __device__ __forceinline__ v16bf load_a_frag(const float* A, int row0,
                                                    int lda, int kbase,
                                                    float scale, int lane) {
  const float* ar = A + (size_t)(row0 + (lane & 15)) * lda + kbase + ((lane >> 4) << 4);
  v16bf a;
#pragma unroll
  for (int e = 0; e < 8; ++e) a[e] = (__bf16)(ar[e] * scale);
#pragma unroll
  for (int e = 0; e < 8; ++e) a[8 + e] = (__bf16)(ar[8 + e] * scale);
  return a;
}

// bf16-native source: two 16B loads, no converts
static __device__ __forceinline__ v16bf load_a_frag_bf(const __bf16* A, int row0,
                                                       int lda, int kbase, int lane) {
  const __bf16* ar = A + (size_t)(row0 + (lane & 15)) * lda + kbase + ((lane >> 4) << 4);
  v8bf lo = *(const v8bf*)ar;         // K block {kh*16 .. kh*16+7}
  v8bf hi = *(const v8bf*)(ar + 8);   // K block {kh*16+8 .. kh*16+15}
  return __builtin_shufflevector(lo, hi, 0, 1, 2, 3, 4, 5, 6, 7,
                                 8, 9, 10, 11, 12, 13, 14, 15);
}

// B: 32x16 (KxN) bf16 with B[k][n] = W[col0+n][k] (row-major W, ld=ldw).
// Lane holds column (lane&15); K = (lane>>4)*16 + e -> 16 consecutive elems.
static __device__ __forceinline__ v16bf load_b_fragW(const float* W, int col0,
                                                     int ldw, int kbase, int lane) {
  const float* wr = W + (size_t)(col0 + (lane & 15)) * ldw + kbase + ((lane >> 4) << 4);
  v16bf b;
#pragma unroll
  for (int e = 0; e < 16; ++e) b[e] = (__bf16)wr[e];
  return b;
}

static __device__ __forceinline__ v16bf load_b_fragW_bf(const __bf16* W, int col0,
                                                        int ldw, int kbase, int lane) {
  const __bf16* wr = W + (size_t)(col0 + (lane & 15)) * ldw + kbase + ((lane >> 4) << 4);
  return *(const v16bf*)wr;  // 32B (address is 32B-aligned by construction)
}

// ---------------- 0a: zero small accumulator region ----------------
__global__ void zero_kernel(unsigned* p, int n) {
  int i = blockIdx.x * blockDim.x + threadIdx.x;
  if (i < n) p[i] = 0u;
}

// ---------------- 0b: f32 -> bf16 bulk convert (n divisible by 4) --------
__global__ void cvt_bf16_kernel(const float* src, __bf16* dst, int n4) {
  int i = blockIdx.x * blockDim.x + threadIdx.x;
  if (i >= n4) return;
  float4 v = ((const float4*)src)[i];
  v4bf o;
  o[0] = (__bf16)v.x; o[1] = (__bf16)v.y; o[2] = (__bf16)v.z; o[3] = (__bf16)v.w;
  ((v4bf*)dst)[i] = o;
}

// ---------------- 1: fused Q/K/V GEMM  C = X * W^T + b (bf16-native) -----
// grid (N/16, 1, 3), block 128 (4 waves). Wave w computes cols [64w,64w+64).
__global__ void qkv_gemm(const __bf16* featbf, const __bf16* zbf,
                         const __bf16* Wqbf, const float* bq,
                         const __bf16* Wkbf, const float* bk,
                         const __bf16* Wvbf, const float* bv,
                         float* qb, float* kb, float* vb) {
  int which = blockIdx.z;
  const __bf16* A    = (which == 2) ? zbf : featbf;
  const __bf16* W    = (which == 0) ? Wqbf : (which == 1) ? Wkbf : Wvbf;
  const float*  bias = (which == 0) ? bq : (which == 1) ? bk : bv;
  float*        out  = (which == 0) ? qb : (which == 1) ? kb : vb;

  int lane = threadIdx.x & 31;
  int wave = threadIdx.x >> 5;
  int row0 = blockIdx.x * 16;
  int cbase = wave * 64;

  v8f acc[4] = {};
#pragma unroll 1
  for (int kk = 0; kk < INC_; kk += 32) {
    v16bf a = load_a_frag_bf(A, row0, INC_, kk, lane);
#pragma unroll
    for (int ct = 0; ct < 4; ++ct) {
      v16bf b = load_b_fragW_bf(W, cbase + ct * 16, INC_, kk, lane);
      acc[ct] = __builtin_amdgcn_wmma_f32_16x16x32_bf16(
          false, a, false, b, (short)0, acc[ct], false, false);
    }
  }
  int r = (lane >> 4) * 8;
  int c = lane & 15;
#pragma unroll
  for (int ct = 0; ct < 4; ++ct) {
    int col = cbase + ct * 16 + c;
    float bcol = bias[col];
#pragma unroll
    for (int g = 0; g < 8; ++g)
      out[(size_t)(row0 + r + g) * INC_ + col] = acc[ct][g] + bcol;
  }
}

// ---------------- 2: xd = (x*scale) @ proj^T, rows = (n,h) --------------
// grid (NH/16, 2) [y: 0=q,1=k], block 64 (2 waves: wave = m-tile 0/1).
__global__ void xd_gemm(const float* qb, const float* kb, const float* proj,
                        const float* tau_p, float* qd, float* kd) {
  const float* A = (blockIdx.y == 0) ? qb : kb;
  float* out     = (blockIdx.y == 0) ? qd : kd;
  float scale = rsqrtf(tau_p[0]) * 0.35355339059327373f;  // tau^-1/2 * 64^-1/4

  int lane  = threadIdx.x & 31;
  int ntile = threadIdx.x >> 5;  // 0 or 1
  int row0  = blockIdx.x * 16;
  v8f acc = {};
#pragma unroll 1
  for (int kk = 0; kk < D_; kk += 32) {
    v16bf a = load_a_frag(A, row0, D_, kk, scale, lane);
    int m = ntile * 16 + (lane & 15);
    v16bf b;
    if (m < M_) {
      const float* pr = proj + (size_t)m * D_ + kk + ((lane >> 4) << 4);
#pragma unroll
      for (int e = 0; e < 16; ++e) b[e] = (__bf16)pr[e];
    } else {
#pragma unroll
      for (int e = 0; e < 16; ++e) b[e] = (__bf16)0.0f;
    }
    acc = __builtin_amdgcn_wmma_f32_16x16x32_bf16(
        false, a, false, b, (short)0, acc, false, false);
  }
  int r = (lane >> 4) * 8, c = lane & 15;
#pragma unroll
  for (int g = 0; g < 8; ++g)
    out[(size_t)(row0 + r + g) * MP_ + ntile * 16 + c] = acc[g];
}

// ---------------- 3: per-head global max of key xd ----------------
__global__ void kmax_kernel(const float* kd, unsigned* khmax, int NH) {
  int r = blockIdx.x * blockDim.x + threadIdx.x;
  if (r >= NH) return;
  float mx = -INFINITY;
  for (int m = 0; m < M_; ++m) mx = fmaxf(mx, kd[(size_t)r * MP_ + m]);
  atomicMax(&khmax[r & (H_ - 1)], fenc(mx));
}

// ---------------- 4: phi = ratio*(exp(xd - diag - mx) + eps) ------------
// In-place over xd buffer (read row, then write). is_query picks row-max.
__global__ void phi_kernel(const float* xbuf, const float* xd_in, float* phi,
                           const float* tau_p, const unsigned* khmax,
                           int NH, int is_query) {
  int r = blockIdx.x * blockDim.x + threadIdx.x;
  if (r >= NH) return;
  float scale = rsqrtf(tau_p[0]) * 0.35355339059327373f;
  const float* xr = xbuf + (size_t)r * D_;
  float ss = 0.f;
#pragma unroll 4
  for (int d = 0; d < D_; ++d) { float x = xr[d] * scale; ss += x * x; }
  float diag = 0.5f * ss;
  float xdv[M_];
  const float* xdr = xd_in + (size_t)r * MP_;
  float mx;
  if (is_query) {
    mx = -INFINITY;
    for (int m = 0; m < M_; ++m) { xdv[m] = xdr[m]; mx = fmaxf(mx, xdv[m]); }
  } else {
    for (int m = 0; m < M_; ++m) xdv[m] = xdr[m];
    mx = fdec(khmax[r & (H_ - 1)]);
  }
  const float ratio = 0.18257418583505536f;  // 1/sqrt(30)
  float* pr = phi + (size_t)r * MP_;
  for (int m = 0; m < M_; ++m)
    pr[m] = ratio * (expf(xdv[m] - diag - mx) + 1e-6f);
  pr[M_] = 0.f;          // zero the pad features so they drop out of all dots
  pr[M_ + 1] = 0.f;
}

// ---------------- 5: kvs[h,m,d] = sum_n kphi*v ; ksum[h,m] = sum_n kphi --
// grid (40, H), block 256. Thread owns d=(t&63), m in {t>>6 + 4j}.
__global__ void kvs_reduce(const float* kphi, const float* vb,
                           float* kvs, float* ksum, int N, int chunk) {
  int h  = blockIdx.y;
  int n0 = blockIdx.x * chunk;
  int n1 = n0 + chunk; if (n1 > N) n1 = N;
  int t  = threadIdx.x;
  int d  = t & 63;
  int mb = t >> 6;  // 0..3
  float acc[8] = {};
  float ks[8]  = {};
  for (int n = n0; n < n1; ++n) {
    float vv = vb[(size_t)n * INC_ + h * D_ + d];
    const float* kr = kphi + ((size_t)n * H_ + h) * MP_;
#pragma unroll
    for (int j = 0; j < 8; ++j) {
      float kv = kr[mb + 4 * j];
      acc[j] += kv * vv;
      if (d == 0) ks[j] += kv;
    }
  }
#pragma unroll
  for (int j = 0; j < 8; ++j) {
    atomicAdd(&kvs[((size_t)h * MP_ + mb + 4 * j) * D_ + d], acc[j]);
    if (d == 0) atomicAdd(&ksum[h * MP_ + mb + 4 * j], ks[j]);
  }
}

// ---------------- 6: z_next = (qphi @ kvs) / (qphi @ ksum) ----------------
__global__ void attn_combine(const float* qphi, const float* kvs,
                             const float* ksum, float* znext, int total) {
  int i = blockIdx.x * blockDim.x + threadIdx.x;
  if (i >= total) return;
  int n = i >> 8;
  int c = i & 255;
  int h = c >> 6;
  int d = c & 63;
  const float* qr  = qphi + ((size_t)n * H_ + h) * MP_;
  const float* kv  = kvs  + (size_t)h * MP_ * D_ + d;
  const float* kss = ksum + h * MP_;
  float num = 0.f, den = 0.f;
#pragma unroll
  for (int m = 0; m < MP_; ++m) {
    float ph = qr[m];
    num += ph * kv[(size_t)m * D_];
    den += ph * kss[m];
  }
  znext[i] = num / den;
}

// ---------------- 7: degree histograms ----------------
__global__ void degree_kernel(const int* ei, unsigned* din, unsigned* dout, int E) {
  int e = blockIdx.x * blockDim.x + threadIdx.x;
  if (e >= E) return;
  atomicAdd(&dout[ei[e]], 1u);      // row -> out-degree
  atomicAdd(&din[ei[E + e]], 1u);   // col -> in-degree
}

// ---------------- 8: edge weights w = 1/sqrt(din[col]*dout[row]) ----------
__global__ void edgew_kernel(const int* ei, const unsigned* din,
                             const unsigned* dout, float* ew, int E) {
  int e = blockIdx.x * blockDim.x + threadIdx.x;
  if (e >= E) return;
  // endpoints of an edge are always counted by that edge -> degrees >= 1
  ew[e] = rsqrtf((float)din[ei[E + e]]) * rsqrtf((float)dout[ei[e]]);
}

// ---------------- 9: scatter z_next[col] += sig(b_h)*w*v[row] -------------
// grid E, block 256 (one edge per block, 256 channels).
__global__ void scatter_kernel(const int* ei, const float* ew, const float* vb,
                               const float* bparam, float* znext, int E) {
  int e = blockIdx.x;
  int c = threadIdx.x;
  int h = c >> 6;
  float sig = 1.f / (1.f + expf(-bparam[h]));
  int row = ei[e], col = ei[E + e];
  float w = ew[e] * sig;
  atomicAdd(&znext[(size_t)col * INC_ + c], w * vb[(size_t)row * INC_ + c]);
}

// ---------------- 10: out = z_next @ Wo^T + bo ----------------
// grid N/16, block 128 (4 waves = 4 col tiles of the 64 output cols).
__global__ void out_gemm(const float* znext, const float* Wo, const float* bo,
                         float* out) {
  int lane = threadIdx.x & 31;
  int wave = threadIdx.x >> 5;
  int row0 = blockIdx.x * 16;
  int col0 = wave * 16;
  v8f acc = {};
#pragma unroll 1
  for (int kk = 0; kk < INC_; kk += 32) {
    v16bf a = load_a_frag(znext, row0, INC_, kk, 1.0f, lane);
    v16bf b = load_b_fragW(Wo, col0, INC_, kk, lane);
    acc = __builtin_amdgcn_wmma_f32_16x16x32_bf16(
        false, a, false, b, (short)0, acc, false, false);
  }
  int r = (lane >> 4) * 8, c = lane & 15;
  float bcol = bo[col0 + c];
#pragma unroll
  for (int g = 0; g < 8; ++g)
    out[(size_t)(row0 + r + g) * D_ + col0 + c] = acc[g] + bcol;
}

extern "C" void kernel_launch(void* const* d_in, const int* in_sizes, int n_in,
                              void* d_out, int out_size, void* d_ws, size_t ws_size,
                              hipStream_t stream) {
  const float* z      = (const float*)d_in[0];
  const float* feat   = (const float*)d_in[1];
  const int*   ei     = (const int*)d_in[2];
  const float* tau    = (const float*)d_in[3];
  const float* Wq     = (const float*)d_in[4];
  const float* bq     = (const float*)d_in[5];
  const float* Wk     = (const float*)d_in[6];
  const float* bk     = (const float*)d_in[7];
  const float* Wv     = (const float*)d_in[8];
  const float* bv     = (const float*)d_in[9];
  const float* Wo     = (const float*)d_in[10];
  const float* bo     = (const float*)d_in[11];
  const float* bparam = (const float*)d_in[12];
  const float* proj   = (const float*)d_in[13];

  const int N  = in_sizes[0] / INC_;  // 20000 (divisible by 16)
  const int E  = in_sizes[2] / 2;     // 320000
  const int NH = N * H_;              // 80000

  // ---- workspace layout (overlaid): ~84 MB high-water ----
  char* ws = (char*)d_ws;
  size_t off = 0;
  auto alloc = [&](size_t bytes) {
    char* p = ws + off;
    off = (off + bytes + 255) & ~(size_t)255;
    return p;
  };
  float* qb   = (float*)alloc((size_t)N * INC_ * 4);   // q; later reused as z_next
  float* kbuf = (float*)alloc((size_t)N * INC_ * 4);   // k
  float* vb   = (float*)alloc((size_t)N * INC_ * 4);   // v
  float* qd   = (float*)alloc((size_t)NH * MP_ * 4);   // feat_bf, then q xd / qphi
  float* kd   = (float*)alloc((size_t)NH * MP_ * 4);   // z_bf,    then k xd / kphi
  float* ew   = (float*)alloc((size_t)E * 4);          // W*_bf,   then edge weights
  int smalln = 2 * N + H_ * MP_ * D_ + H_ * MP_ + H_;  // din,dout,kvs,ksum,khmax
  unsigned* small = (unsigned*)alloc((size_t)smalln * 4);
  unsigned* din   = small;
  unsigned* dout  = small + N;
  float*    kvs   = (float*)(small + 2 * N);
  float*    ksum  = kvs + H_ * MP_ * D_;
  unsigned* khmax = (unsigned*)(ksum + H_ * MP_);
  float* znext = qb;
  float* qphi  = qd;
  float* kphi  = kd;

  // bf16 staging overlays (dead once qkv_gemm has run):
  //   feat_bf/z_bf: N*256 bf16 = 10.24 MB each, fits exactly in qd/kd (10.24 MB)
  //   Wq/Wk/Wv bf16: 128 KB each, packed into ew (1.28 MB)
  __bf16* featbf = (__bf16*)qd;
  __bf16* zbf    = (__bf16*)kd;
  __bf16* Wqbf   = (__bf16*)ew;
  __bf16* Wkbf   = Wqbf + (size_t)INC_ * INC_;
  __bf16* Wvbf   = Wkbf + (size_t)INC_ * INC_;

  zero_kernel<<<(smalln + 255) / 256, 256, 0, stream>>>(small, smalln);

  int nin4 = N * INC_ / 4;
  int nw4  = INC_ * INC_ / 4;
  cvt_bf16_kernel<<<(nin4 + 255) / 256, 256, 0, stream>>>(feat, featbf, nin4);
  cvt_bf16_kernel<<<(nin4 + 255) / 256, 256, 0, stream>>>(z, zbf, nin4);
  cvt_bf16_kernel<<<(nw4 + 255) / 256, 256, 0, stream>>>(Wq, Wqbf, nw4);
  cvt_bf16_kernel<<<(nw4 + 255) / 256, 256, 0, stream>>>(Wk, Wkbf, nw4);
  cvt_bf16_kernel<<<(nw4 + 255) / 256, 256, 0, stream>>>(Wv, Wvbf, nw4);

  qkv_gemm<<<dim3(N / 16, 1, 3), 128, 0, stream>>>(featbf, zbf, Wqbf, bq,
                                                   Wkbf, bk, Wvbf, bv,
                                                   qb, kbuf, vb);
  xd_gemm<<<dim3(NH / 16, 2), 64, 0, stream>>>(qb, kbuf, proj, tau, qd, kd);
  kmax_kernel<<<(NH + 255) / 256, 256, 0, stream>>>(kd, khmax, NH);
  phi_kernel<<<(NH + 255) / 256, 256, 0, stream>>>(qb, qd, qphi, tau, khmax, NH, 1);
  phi_kernel<<<(NH + 255) / 256, 256, 0, stream>>>(kbuf, kd, kphi, tau, khmax, NH, 0);
  int chunk = (N + 39) / 40;
  kvs_reduce<<<dim3(40, H_), 256, 0, stream>>>(kphi, vb, kvs, ksum, N, chunk);
  attn_combine<<<(N * INC_ + 255) / 256, 256, 0, stream>>>(qphi, kvs, ksum,
                                                           znext, N * INC_);
  degree_kernel<<<(E + 255) / 256, 256, 0, stream>>>(ei, din, dout, E);
  edgew_kernel<<<(E + 255) / 256, 256, 0, stream>>>(ei, din, dout, ew, E);
  scatter_kernel<<<E, 256, 0, stream>>>(ei, ew, vb, bparam, znext, E);
  out_gemm<<<N / 16, 128, 0, stream>>>(znext, Wo, bo, (float*)d_out);
}
